// PAWSA_44444321579633
// MI455X (gfx1250) — matrix operand order repaired and checked by reading, not compiled, not run on hardware
//
#include <hip/hip_runtime.h>
#include <hip/hip_bf16.h>
#include <math.h>

typedef __attribute__((ext_vector_type(16))) _Float16 v16h;
typedef __attribute__((ext_vector_type(8)))  _Float16 h8;
typedef __attribute__((ext_vector_type(4)))  _Float16 h4;
typedef __attribute__((ext_vector_type(2)))  _Float16 h2;
typedef __attribute__((ext_vector_type(8)))  float    v8f;
typedef __attribute__((ext_vector_type(4)))  float    f4;

#define TPB   128
#define NTOK  49
#define NPRI  3
#define NEXT  52      // 49 + 3
#define CCH   96
#define NHEAD 3
#define HD    32
#define NW    1024
#define NWIN  8192
#define QKVW_H (96 * 288)   // 27648 halfs
#define PRJW_H (96 * 96)    //  9216 halfs

__device__ __forceinline__ v8f wmma_f16(v16h a, v16h b, v8f c) {
  // D = A(16x32 f16) x B(32x16 f16) + C(16x16 f32)
  return __builtin_amdgcn_wmma_f32_16x16x32_f16(false, a, false, b, (short)0, c, false, false);
}

__device__ __forceinline__ v8f zero8() {
  v8f z;
#pragma unroll
  for (int r = 0; r < 8; ++r) z[r] = 0.0f;
  return z;   // zero C lowers to the inline-0 SRC2 WMMA form (no splat movs)
}

// A-operand (16x32, MxK) from row-major f16 (K contiguous), contiguous k order, same as load_B16:
// lane l: m=l&15, hi=l>>4 ; halves h -> K = 16*hi+h  (16 consecutive halfs)
__device__ __forceinline__ v16h load_A16(const _Float16* src, int ld, int lane) {
  int m = lane & 15, hi = lane >> 4;
  const _Float16* p = src + m * ld + 16 * hi;    // 16B aligned by construction
  h8 lo = *(const h8*)(p);
  h8 hh = *(const h8*)(p + 8);
  return __builtin_shufflevector(lo, hh, 0, 1, 2, 3, 4, 5, 6, 7,
                                         8, 9, 10, 11, 12, 13, 14, 15);
}

// B-operand (32x16, KxN): source row-major by N with K contiguous (src[n][k]):
// lane l: n=l&15 ; halves h -> K = 16*(l>>4)+h  (16 consecutive halfs)
__device__ __forceinline__ v16h load_B16(const _Float16* src, int ld, int lane) {
  int n = lane & 15, hi = lane >> 4;
  const _Float16* p = src + n * ld + 16 * hi;    // 16B aligned by construction
  h8 lo = *(const h8*)(p);
  h8 hh = *(const h8*)(p + 8);
  return __builtin_shufflevector(lo, hh, 0, 1, 2, 3, 4, 5, 6, 7,
                                         8, 9, 10, 11, 12, 13, 14, 15);
}

// Fallback: B operand straight from row-major f32 weights (converted per use).
__device__ __forceinline__ v16h load_B_gw(const float* w, int ld, int lane) {
  int n = lane & 15, hi = lane >> 4;
  const float* p0 = w + n * ld + 16 * hi;
  v16h b;
#pragma unroll
  for (int j = 0; j < 16; ++j) b[j] = (_Float16)p0[j];
  return b;
}

template <bool W16>
__global__ __launch_bounds__(TPB) void pawsa_fused(
    const float* __restrict__ x, const float* __restrict__ mask,
    const float* __restrict__ uk, const float* __restrict__ fg,
    const float* __restrict__ bg, const float* __restrict__ qkv_w,
    const float* __restrict__ qkv_b, const float* __restrict__ proj_w,
    const float* __restrict__ proj_b, const float* __restrict__ bt,
    const _Float16* __restrict__ qkv_w16, const _Float16* __restrict__ proj_w16,
    float* __restrict__ out)
{
  // LDS arena (65,344 B):
  //   xe/yb : [64][96] f16  (x_ext staging, later attention output)  @ 0
  //   qf    : [50][96] f16  (rows >=49 clamp to dump row 49)         @ 12288
  //   kf    : [53][96] f16  (rows >=52 clamp to dump row 52)         @ 21888
  //   vT    : [96][64] f16  (v transposed, token minor, all written) @ 32064
  //   sc    : [50][64] f32  (scores; dump row 49)                    @ 44352
  //   pp    : [64][64] f16  (probabilities, A-operand-ready)         @ 57152
  //   po    : [50][96] f32  (proj staging; overlays sc+pp)           @ 44352
  __shared__ __align__(16) char smem[65344];
  _Float16* xe = (_Float16*)(smem);
  _Float16* yb = (_Float16*)(smem);
  _Float16* qf = (_Float16*)(smem + 12288);
  _Float16* kf = (_Float16*)(smem + 21888);
  _Float16* vT = (_Float16*)(smem + 32064);
  float*    sc = (float*)   (smem + 44352);
  _Float16* pp = (_Float16*)(smem + 57152);
  float*    po = (float*)   (smem + 44352);

  const int tid  = threadIdx.x;
  const int lane = tid & 31;
  const int wv   = tid >> 5;          // 4 waves
  const int n16  = lane & 15;
  const int hi   = lane >> 4;
  const int win  = blockIdx.x;        // window index (B_)
  const int wmsk = win & (NW - 1);    // mask window id
  const int bb   = win >> 10;         // batch id
  const float scale = 0.17677669529663687f;  // hd^-0.5

  __builtin_prefetch(W16 ? (const void*)qkv_w16 : (const void*)qkv_w, 0, 0);

  // ---- Phase 0: stage x_ext as f16 via float4 loads; rows 52..63 zeroed
  for (int e = tid; e < 64 * 24; e += TPB) {
    int row = e / 24, q = e - row * 24;
    f4 v;
    v[0] = 0.f; v[1] = 0.f; v[2] = 0.f; v[3] = 0.f;
    if (row < NTOK) {
      v = *(const f4*)(x + (size_t)win * NTOK * CCH + row * CCH + q * 4);
    } else if (row < NEXT) {
      const float* pr = (row == NTOK) ? uk : (row == NTOK + 1) ? fg : bg;
      v = *(const f4*)(pr + bb * CCH + q * 4);
    }
    h4 o;
    o[0] = (_Float16)v[0]; o[1] = (_Float16)v[1];
    o[2] = (_Float16)v[2]; o[3] = (_Float16)v[3];
    *(h4*)(xe + row * CCH + q * 4) = o;
  }
  __syncthreads();

  // ---- Phase 1: QKV GEMM  (x_ext[52x96] @ qkv_w^T[96x288]; bias added at store)
  {
    const int mt = wv;  // one 16-row M-tile per wave
    v16h a0 = load_A16(xe + mt * 16 * CCH + 0,  CCH, lane);
    v16h a1 = load_A16(xe + mt * 16 * CCH + 32, CCH, lane);
    v16h a2 = load_A16(xe + mt * 16 * CCH + 64, CCH, lane);
    for (int nt = 0; nt < 18; ++nt) {
      const int cbase = nt * 16;
      const float bv = qkv_b[cbase + n16];
      v8f acc = zero8();
      v16h b0, b1, b2;
      if constexpr (W16) {
        b0 = load_B16(qkv_w16 + (size_t)cbase * CCH + 0,  CCH, lane);
        b1 = load_B16(qkv_w16 + (size_t)cbase * CCH + 32, CCH, lane);
        b2 = load_B16(qkv_w16 + (size_t)cbase * CCH + 64, CCH, lane);
      } else {
        b0 = load_B_gw(qkv_w + (size_t)cbase * CCH + 0,  CCH, lane);
        b1 = load_B_gw(qkv_w + (size_t)cbase * CCH + 32, CCH, lane);
        b2 = load_B_gw(qkv_w + (size_t)cbase * CCH + 64, CCH, lane);
      }
      acc = wmma_f16(a0, b0, acc);
      acc = wmma_f16(a1, b1, acc);
      acc = wmma_f16(a2, b2, acc);

      if (cbase < 96) {                     // Q -> qf (dump row 49)
#pragma unroll
        for (int r = 0; r < 8; ++r) {
          int m = mt * 16 + r + 8 * hi;
          int mc = m < NTOK ? m : NTOK;
          qf[mc * CCH + cbase + n16] = (_Float16)(acc[r] + bv);
        }
      } else if (cbase < 192) {             // K -> kf (dump row 52)
#pragma unroll
        for (int r = 0; r < 8; ++r) {
          int m = mt * 16 + r + 8 * hi;
          int mc = m < NEXT ? m : NEXT;
          kf[mc * CCH + (cbase - 96) + n16] = (_Float16)(acc[r] + bv);
        }
      } else {                              // V -> vT transposed, packed b32 stores
        int c = cbase - 192 + n16;
        _Float16* vp = vT + c * 64 + mt * 16 + 8 * hi;
#pragma unroll
        for (int r = 0; r < 8; r += 2) {
          h2 t;
          t[0] = (_Float16)(acc[r] + bv);
          t[1] = (_Float16)(acc[r + 1] + bv);
          *(h2*)(vp + r) = t;
        }
      }
    }
  }
  __syncthreads();

  // ---- Phases 2..4 per head: scores -> softmax -> attn@V
  for (int h = 0; h < NHEAD; ++h) {
    // scores: q[49x32] @ k^T[32x52]  (K = hd = 32: one WMMA step)
    {
      const int mt = wv;
      v16h a = load_A16(qf + mt * 16 * CCH + h * HD, CCH, lane);
#pragma unroll
      for (int nt = 0; nt < 4; ++nt) {
        v16h b = load_B16(kf + nt * 16 * CCH + h * HD, CCH, lane);
        v8f acc = zero8();
        acc = wmma_f16(a, b, acc);
#pragma unroll
        for (int r = 0; r < 8; ++r) {
          int m = mt * 16 + r + 8 * hi;
          int mc = m < NTOK ? m : NTOK;      // dump row 49
          sc[mc * 64 + nt * 16 + n16] = acc[r];
        }
      }
    }
    __syncthreads();

    // softmax over 52 keys (f32), + rel-pos bias + window mask; emit f16 probs
    if (tid < NTOK) {
      const int m = tid;
      const int mh = m / 7, mw = m - mh * 7;
      const float* mrow = mask + (size_t)wmsk * NTOK * NTOK + m * NTOK;
      float mx = -1e30f;
      for (int n = 0; n < NEXT; ++n) {
        float s = sc[m * 64 + n] * scale;
        int idx;
        if (n < NTOK) {
          int nh = n / 7, nw = n - nh * 7;
          idx = (mh - nh + 6) * 13 + (mw - nw + 6);
          s += mrow[n];
        } else {
          idx = 169 + (n - NTOK);
        }
        s += bt[idx * NHEAD + h];
        sc[m * 64 + n] = s;
        mx = fmaxf(mx, s);
      }
      float sum = 0.0f;
      for (int n = 0; n < NEXT; ++n) {
        float e = __expf(sc[m * 64 + n] - mx);
        sc[m * 64 + n] = e;
        sum += e;
      }
      float inv = 1.0f / sum;
      _Float16* prow = pp + m * 64;
#pragma unroll
      for (int n = 0; n < 64; n += 2) {     // paired b32 stores
        h2 t;
        t[0] = (n + 0 < NEXT) ? (_Float16)(sc[m * 64 + n + 0] * inv) : (_Float16)0.0f;
        t[1] = (n + 1 < NEXT) ? (_Float16)(sc[m * 64 + n + 1] * inv) : (_Float16)0.0f;
        *(h2*)(prow + n) = t;
      }
    }
    __syncthreads();

    // attn @ V : p[49x64(pad)] @ v[64x32] -> y[:, h*32 .. h*32+31]
    {
      const int mt = wv;
      v16h a0 = load_A16(pp + mt * 16 * 64 + 0,  64, lane);
      v16h a1 = load_A16(pp + mt * 16 * 64 + 32, 64, lane);
#pragma unroll
      for (int nt = 0; nt < 2; ++nt) {
        v8f acc = zero8();
        v16h b0 = load_B16(vT + (h * HD + nt * 16) * 64 + 0,  64, lane);
        v16h b1 = load_B16(vT + (h * HD + nt * 16) * 64 + 32, 64, lane);
        acc = wmma_f16(a0, b0, acc);
        acc = wmma_f16(a1, b1, acc);
#pragma unroll
        for (int r = 0; r < 8; ++r)
          yb[(mt * 16 + r + 8 * hi) * CCH + h * HD + nt * 16 + n16] = (_Float16)acc[r];
      }
    }
    __syncthreads();
  }

  // ---- Phase 5: output projection  y[49x96] @ proj_w^T[96x96] + proj_b
  {
    const int mt = wv;
    v16h a0 = load_A16(yb + mt * 16 * CCH + 0,  CCH, lane);
    v16h a1 = load_A16(yb + mt * 16 * CCH + 32, CCH, lane);
    v16h a2 = load_A16(yb + mt * 16 * CCH + 64, CCH, lane);
#pragma unroll
    for (int nt = 0; nt < 6; ++nt) {
      const float bv = proj_b[nt * 16 + n16];
      v8f acc = zero8();
      v16h b0, b1, b2;
      if constexpr (W16) {
        b0 = load_B16(proj_w16 + (size_t)(nt * 16) * CCH + 0,  CCH, lane);
        b1 = load_B16(proj_w16 + (size_t)(nt * 16) * CCH + 32, CCH, lane);
        b2 = load_B16(proj_w16 + (size_t)(nt * 16) * CCH + 64, CCH, lane);
      } else {
        b0 = load_B_gw(proj_w + (size_t)(nt * 16) * CCH + 0,  CCH, lane);
        b1 = load_B_gw(proj_w + (size_t)(nt * 16) * CCH + 32, CCH, lane);
        b2 = load_B_gw(proj_w + (size_t)(nt * 16) * CCH + 64, CCH, lane);
      }
      acc = wmma_f16(a0, b0, acc);
      acc = wmma_f16(a1, b1, acc);
      acc = wmma_f16(a2, b2, acc);
#pragma unroll
      for (int r = 0; r < 8; ++r) {
        int m = mt * 16 + r + 8 * hi;
        int mc = m < NTOK ? m : NTOK;        // dump row 49 of po
        po[mc * CCH + nt * 16 + n16] = acc[r] + bv;
      }
    }
  }
  __syncthreads();

  // coalesced float4 writeback of the 49 real rows
  {
    float* dst = out + (size_t)win * NTOK * CCH;
    for (int e = tid; e < (NTOK * CCH) / 4; e += TPB)
      ((f4*)dst)[e] = ((const f4*)po)[e];
  }
}

// ---- one-shot f32 -> f16 weight conversion into workspace
__global__ __launch_bounds__(256) void pawsa_cvt_w(
    const float* __restrict__ qkv_w, const float* __restrict__ proj_w,
    _Float16* __restrict__ w16)
{
  int i = blockIdx.x * 256 + threadIdx.x;
  if (i < QKVW_H) w16[i] = (_Float16)qkv_w[i];
  else if (i < QKVW_H + PRJW_H) w16[i] = (_Float16)proj_w[i - QKVW_H];
}

// ---- prior highway: v = (prior @ qkv_w^T + qkv_b)[v-part]; out2 = v @ proj_w^T + proj_b
__global__ __launch_bounds__(TPB) void pawsa_prior_hw(
    const float* __restrict__ uk, const float* __restrict__ fg,
    const float* __restrict__ bg, const float* __restrict__ qkv_w,
    const float* __restrict__ qkv_b, const float* __restrict__ proj_w,
    const float* __restrict__ proj_b, float* __restrict__ out2)
{
  __shared__ float pri[24 * 96];
  __shared__ float vv[24 * 96];
  const int tid = threadIdx.x;
  for (int e = tid; e < 24 * 96; e += TPB) {
    int r = e / 96, c = e - (e / 96) * 96;
    int b = r / 3, j = r - (r / 3) * 3;
    const float* src = (j == 0) ? uk : (j == 1) ? fg : bg;
    pri[r * 96 + c] = src[b * 96 + c];
  }
  __syncthreads();
  for (int e = tid; e < 24 * 96; e += TPB) {
    int r = e / 96, k = e - (e / 96) * 96;
    float s = qkv_b[192 + k];
    for (int j = 0; j < 96; ++j) s += pri[r * 96 + j] * qkv_w[(size_t)(192 + k) * 96 + j];
    vv[r * 96 + k] = s;
  }
  __syncthreads();
  for (int e = tid; e < 24 * 96; e += TPB) {
    int r = e / 96, c = e - (e / 96) * 96;
    float s = proj_b[c];
    for (int k = 0; k < 96; ++k) s += vv[r * 96 + k] * proj_w[(size_t)c * 96 + k];
    out2[r * 96 + c] = s;
  }
}

extern "C" void kernel_launch(void* const* d_in, const int* in_sizes, int n_in,
                              void* d_out, int out_size, void* d_ws, size_t ws_size,
                              hipStream_t stream) {
  const float* x      = (const float*)d_in[0];
  const float* mask   = (const float*)d_in[1];
  const float* uk     = (const float*)d_in[2];
  const float* fg     = (const float*)d_in[3];
  const float* bg     = (const float*)d_in[4];
  const float* qkv_w  = (const float*)d_in[5];
  const float* qkv_b  = (const float*)d_in[6];
  const float* proj_w = (const float*)d_in[7];
  const float* proj_b = (const float*)d_in[8];
  const float* bt     = (const float*)d_in[9];
  float* out = (float*)d_out;

  const size_t w16_bytes = (size_t)(QKVW_H + PRJW_H) * sizeof(_Float16);
  if (ws_size >= w16_bytes) {
    _Float16* w16 = (_Float16*)d_ws;
    pawsa_cvt_w<<<dim3((QKVW_H + PRJW_H + 255) / 256), dim3(256), 0, stream>>>(
        qkv_w, proj_w, w16);
    pawsa_fused<true><<<dim3(NWIN), dim3(TPB), 0, stream>>>(
        x, mask, uk, fg, bg, qkv_w, qkv_b, proj_w, proj_b, bt,
        w16, w16 + QKVW_H, out);
  } else {
    pawsa_fused<false><<<dim3(NWIN), dim3(TPB), 0, stream>>>(
        x, mask, uk, fg, bg, qkv_w, qkv_b, proj_w, proj_b, bt,
        (const _Float16*)nullptr, (const _Float16*)nullptr, out);
  }

  float* out2 = out + (size_t)NWIN * NTOK * CCH;
  pawsa_prior_hw<<<dim3(1), dim3(TPB), 0, stream>>>(
      uk, fg, bg, qkv_w, qkv_b, proj_w, proj_b, out2);
}
